// OptimizedECCModel_74594991997477
// MI455X (gfx1250) — compile-verified, matrix operand order checked by reading, not run on hardware
//
#include <hip/hip_runtime.h>
#include <math.h>

// ------------------------------------------------------------------
// ECC GNN for MI455X (gfx1250).  Main compute = two bf16 WMMA GEMMs
// per layer:
//   (1) msg = u @ B, u[e, i*32+k] = h_src[e,i]*hid[e,k] built on the
//       fly per 32-edge tile (rank-1 structure per K-chunk: the hid
//       fragment is loop-invariant, only the h_src scalar varies),
//       B[2112,64] = static reshuffle of e2_w/e2_b (bias folded).
//   (2) h @ self_w, standard [N,64]x[64,64] GEMM.
// Edge GEMM: 32-edge x 64-col tile per wave, software double-buffered
// B fragments, bias chunks peeled into an epilogue.
// __launch_bounds__(128,1) gives the full VGPR file (no spills).
// ------------------------------------------------------------------

#define H_DIM    64
#define EH_DIM   32
#define NLAYERS  3
#define IN_DIM   16
#define NGRAPH   8
#define KTOT     2112          // 2048 (bilinear) + 64 (bias rows)
#define NKCH     66            // KTOT / 32
#define BN_EPS   1e-5f
#define EG_WAVES 4
#define EG_ROWS  32            // edges per wave-tile (2 M-subtiles)

typedef __attribute__((ext_vector_type(16))) __bf16 v16bf;
typedef __attribute__((ext_vector_type(8)))  float  v8f;

static __device__ __forceinline__ unsigned short bf16bits(float f) {
  unsigned u = __float_as_uint(f);
  return (unsigned short)((u + 0x7FFFu + ((u >> 16) & 1u)) >> 16);
}
static __device__ __forceinline__ __bf16 bits2bf(unsigned short h) {
  __bf16 r; __builtin_memcpy(&r, &h, sizeof(r)); return r;
}
static __device__ __forceinline__ __bf16 f2bf(float f) {
  return bits2bf(bf16bits(f));
}
static __device__ __forceinline__ v16bf load_frag(const unsigned short* p) {
  return *(const v16bf*)p;   // 32B-aligned by construction
}

// -------------------------- utility -------------------------------
__global__ void zero_kernel(float* p, int n) {
  int i = blockIdx.x * blockDim.x + threadIdx.x;
  if (i < n) p[i] = 0.0f;
}

__global__ void deg_kernel(const int* __restrict__ dst, float* __restrict__ deg, int E) {
  int e = blockIdx.x * blockDim.x + threadIdx.x;
  if (e < E) atomicAdd(&deg[dst[e]], 1.0f);
}

// --------------------- weight pre-swizzle -------------------------
// Bsw layout: idx = (((l*66+kc)*4 + nt)*32 + lane)*16 + j
// fragment element j of lane covers K = kc*32 + off, N = nt*16 + (lane&15)
// off = (j>=8 ? 16:0) + (lane>>4)*8 + (j&7)   [16-bit B 32x16 layout]
__global__ void prep_bsw_kernel(const float* __restrict__ e2w,
                                const float* __restrict__ e2b,
                                unsigned short* __restrict__ Bsw, int total) {
  int idx = blockIdx.x * blockDim.x + threadIdx.x;
  if (idx >= total) return;
  int j    = idx & 15;
  int lane = (idx >> 4) & 31;
  int nt   = (idx >> 9) & 3;
  int t    = idx >> 11;          // l*66 + kc
  int kc   = t % NKCH;
  int l    = t / NKCH;
  int off  = ((j >= 8) ? 16 : 0) + (lane >> 4) * 8 + (j & 7);
  int m    = kc * 32 + off;      // global K index in [0,2112)
  int n    = nt * 16 + (lane & 15);
  float v;
  if (m < 2048) {                        // u element h_src[i]*hid[k]
    int i = m >> 5, k = m & 31;
    v = e2w[((size_t)(l * EH_DIM + k)) * (H_DIM * H_DIM) + i * H_DIM + n];
  } else {                               // bias row: u element h_src[i]
    v = e2b[(size_t)l * (H_DIM * H_DIM) + (m - 2048) * H_DIM + n];
  }
  Bsw[idx] = bf16bits(v);
}

// Wsw layout: idx = (((l*2+kc)*4 + nt)*32 + lane)*16 + j
__global__ void prep_wsw_kernel(const float* __restrict__ self_w,
                                unsigned short* __restrict__ Wsw, int total) {
  int idx = blockIdx.x * blockDim.x + threadIdx.x;
  if (idx >= total) return;
  int j    = idx & 15;
  int lane = (idx >> 4) & 31;
  int nt   = (idx >> 9) & 3;
  int t    = idx >> 11;          // l*2 + kc
  int kc   = t & 1;
  int l    = t >> 1;
  int off  = ((j >= 8) ? 16 : 0) + (lane >> 4) * 8 + (j & 7);
  int K    = kc * 32 + off;
  int n    = nt * 16 + (lane & 15);
  Wsw[idx] = bf16bits(self_w[(size_t)l * (H_DIM * H_DIM) + K * H_DIM + n]);
}

// ------------------------- embedding ------------------------------
__global__ void emb_kernel(const float* __restrict__ x,
                           const float* __restrict__ emb_w,
                           const float* __restrict__ emb_b,
                           float* __restrict__ h,
                           unsigned short* __restrict__ hbf, int N) {
  int idx = blockIdx.x * blockDim.x + threadIdx.x;
  if (idx >= N * H_DIM) return;
  int n = idx >> 6, o = idx & 63;
  float acc = emb_b[o];
#pragma unroll
  for (int i = 0; i < IN_DIM; ++i)
    acc += x[(size_t)n * IN_DIM + i] * emb_w[i * H_DIM + o];
  h[idx]   = acc;
  hbf[idx] = bf16bits(acc);
}

// ------------------ edge attr + edge-net layer 1 ------------------
__global__ void hid_kernel(const float* __restrict__ h,
                           const int* __restrict__ src,
                           const int* __restrict__ dst,
                           const float* __restrict__ e1w,
                           const float* __restrict__ e1b,
                           float* __restrict__ hid, int E) {
  int e = blockIdx.x * blockDim.x + threadIdx.x;
  if (e >= E) return;
  const float* hs = h + (size_t)src[e] * H_DIM;
  const float* hd = h + (size_t)dst[e] * H_DIM;
  float d0 = hs[0] - hd[0], d1 = hs[1] - hd[1], d2 = hs[2] - hd[2];
  float ea = sqrtf(d0 * d0 + d1 * d1 + d2 * d2);
  float* o = hid + (size_t)e * EH_DIM;
#pragma unroll
  for (int k = 0; k < EH_DIM; ++k)
    o[k] = fmaxf(ea * e1w[k] + e1b[k], 0.0f);
}

// ------------------- fused edge-weight GEMM -----------------------
// One wave = one 32-edge x 64-output tile; 66 K-chunks of 32.
// hid fragment is K-loop-invariant -> hoisted in bf16; B fragments
// software double-buffered; bias chunks peeled.
__global__ void __launch_bounds__(EG_WAVES * 32, 1)
edge_gemm_kernel(const float* __restrict__ h,
                 const float* __restrict__ hid,
                 const unsigned short* __restrict__ Bsw, // layer slice
                 const int* __restrict__ src,
                 const int* __restrict__ dst,
                 float* __restrict__ agg, int E) {
  __shared__ float hsrcL[EG_WAVES][EG_ROWS][H_DIM];
  __shared__ float hidL [EG_WAVES][EG_ROWS][EH_DIM];
  __shared__ int   srcL [EG_WAVES][EG_ROWS];
  __shared__ int   dstL [EG_WAVES][EG_ROWS];
  const int wave  = threadIdx.x >> 5;
  const int lane  = threadIdx.x & 31;
  const int tile  = blockIdx.x * EG_WAVES + wave;
  const int ebase = tile * EG_ROWS;
  const bool active = ebase < E;

  if (active) {
    int e = ebase + lane;                 // one edge per lane (EG_ROWS==32)
    srcL[wave][lane] = (e < E) ? src[e] : 0;
    dstL[wave][lane] = (e < E) ? dst[e] : 0;
  }
  __syncthreads();
  if (active) {
    for (int idx = lane; idx < EG_ROWS * H_DIM; idx += 32) {
      int r = idx >> 6, c = idx & 63;
      hsrcL[wave][r][c] = h[(size_t)srcL[wave][r] * H_DIM + c];
    }
    for (int idx = lane; idx < EG_ROWS * EH_DIM; idx += 32) {
      int r = idx >> 5, k = idx & 31;
      int ge = ebase + r;
      hidL[wave][r][k] = (ge < E) ? hid[(size_t)ge * EH_DIM + k] : 0.0f;
    }
  }
  __syncthreads();
  if (!active) return;

  const int row  = lane & 15;
  const int half = lane >> 4;
  const float* hs0 = &hsrcL[wave][row][0];
  const float* hs1 = &hsrcL[wave][16 + row][0];
  const float* hi0 = &hidL[wave][row][0];
  const float* hi1 = &hidL[wave][16 + row][0];

  // Loop-invariant hid fragments in bf16 (A layout, 16-bit 16x32).
  v16bf hf0, hf1;
#pragma unroll
  for (int j = 0; j < 16; ++j) {
    int off = ((j >= 8) ? 16 : 0) + half * 8 + (j & 7);
    hf0[j] = f2bf(hi0[off]);
    hf1[j] = f2bf(hi1[off]);
  }

  v8f A00 = {}, A01 = {}, A02 = {}, A03 = {};   // edges 0..15  x 4 N-tiles
  v8f A10 = {}, A11 = {}, A12 = {}, A13 = {};   // edges 16..31 x 4 N-tiles

  // ---- steady state: kc = 0..63, u = s * hid ----
  const size_t lframe = (size_t)lane * 16;
  v16bf b0 = load_frag(Bsw + lframe);
  v16bf b1 = load_frag(Bsw + 512 + lframe);
  v16bf b2 = load_frag(Bsw + 1024 + lframe);
  v16bf b3 = load_frag(Bsw + 1536 + lframe);
#pragma unroll 1
  for (int kc = 0; kc < 64; ++kc) {
    // prefetch next chunk's B fragments (kc+1 <= 64 < NKCH, always valid)
    const unsigned short* np = Bsw + (size_t)(kc + 1) * 2048 + lframe;
    v16bf n0 = load_frag(np);
    v16bf n1 = load_frag(np + 512);
    v16bf n2 = load_frag(np + 1024);
    v16bf n3 = load_frag(np + 1536);

    __bf16 s0 = f2bf(hs0[kc]);
    __bf16 s1 = f2bf(hs1[kc]);
    v16bf a0, a1;
#pragma unroll
    for (int j = 0; j < 16; ++j) {
      a0[j] = (__bf16)(hf0[j] * s0);
      a1[j] = (__bf16)(hf1[j] * s1);
    }
    A00 = __builtin_amdgcn_wmma_f32_16x16x32_bf16(false, a0, false, b0, (short)0, A00, false, false);
    A10 = __builtin_amdgcn_wmma_f32_16x16x32_bf16(false, a1, false, b0, (short)0, A10, false, false);
    A01 = __builtin_amdgcn_wmma_f32_16x16x32_bf16(false, a0, false, b1, (short)0, A01, false, false);
    A11 = __builtin_amdgcn_wmma_f32_16x16x32_bf16(false, a1, false, b1, (short)0, A11, false, false);
    A02 = __builtin_amdgcn_wmma_f32_16x16x32_bf16(false, a0, false, b2, (short)0, A02, false, false);
    A12 = __builtin_amdgcn_wmma_f32_16x16x32_bf16(false, a1, false, b2, (short)0, A12, false, false);
    A03 = __builtin_amdgcn_wmma_f32_16x16x32_bf16(false, a0, false, b3, (short)0, A03, false, false);
    A13 = __builtin_amdgcn_wmma_f32_16x16x32_bf16(false, a1, false, b3, (short)0, A13, false, false);
    b0 = n0; b1 = n1; b2 = n2; b3 = n3;
  }

  // ---- epilogue: bias chunks kc = 64 (in b*), kc = 65 ----
  {
    const unsigned short* p65 = Bsw + (size_t)65 * 2048 + lframe;
    v16bf c0 = load_frag(p65);
    v16bf c1 = load_frag(p65 + 512);
    v16bf c2 = load_frag(p65 + 1024);
    v16bf c3 = load_frag(p65 + 1536);
    v16bf a0, a1;
#pragma unroll
    for (int j = 0; j < 16; ++j) {     // u = h_src[0..31]
      int off = ((j >= 8) ? 16 : 0) + half * 8 + (j & 7);
      a0[j] = f2bf(hs0[off]);
      a1[j] = f2bf(hs1[off]);
    }
    A00 = __builtin_amdgcn_wmma_f32_16x16x32_bf16(false, a0, false, b0, (short)0, A00, false, false);
    A10 = __builtin_amdgcn_wmma_f32_16x16x32_bf16(false, a1, false, b0, (short)0, A10, false, false);
    A01 = __builtin_amdgcn_wmma_f32_16x16x32_bf16(false, a0, false, b1, (short)0, A01, false, false);
    A11 = __builtin_amdgcn_wmma_f32_16x16x32_bf16(false, a1, false, b1, (short)0, A11, false, false);
    A02 = __builtin_amdgcn_wmma_f32_16x16x32_bf16(false, a0, false, b2, (short)0, A02, false, false);
    A12 = __builtin_amdgcn_wmma_f32_16x16x32_bf16(false, a1, false, b2, (short)0, A12, false, false);
    A03 = __builtin_amdgcn_wmma_f32_16x16x32_bf16(false, a0, false, b3, (short)0, A03, false, false);
    A13 = __builtin_amdgcn_wmma_f32_16x16x32_bf16(false, a1, false, b3, (short)0, A13, false, false);
#pragma unroll
    for (int j = 0; j < 16; ++j) {     // u = h_src[32..63]
      int off = ((j >= 8) ? 16 : 0) + half * 8 + (j & 7);
      a0[j] = f2bf(hs0[32 + off]);
      a1[j] = f2bf(hs1[32 + off]);
    }
    A00 = __builtin_amdgcn_wmma_f32_16x16x32_bf16(false, a0, false, c0, (short)0, A00, false, false);
    A10 = __builtin_amdgcn_wmma_f32_16x16x32_bf16(false, a1, false, c0, (short)0, A10, false, false);
    A01 = __builtin_amdgcn_wmma_f32_16x16x32_bf16(false, a0, false, c1, (short)0, A01, false, false);
    A11 = __builtin_amdgcn_wmma_f32_16x16x32_bf16(false, a1, false, c1, (short)0, A11, false, false);
    A02 = __builtin_amdgcn_wmma_f32_16x16x32_bf16(false, a0, false, c2, (short)0, A02, false, false);
    A12 = __builtin_amdgcn_wmma_f32_16x16x32_bf16(false, a1, false, c2, (short)0, A12, false, false);
    A03 = __builtin_amdgcn_wmma_f32_16x16x32_bf16(false, a0, false, c3, (short)0, A03, false, false);
    A13 = __builtin_amdgcn_wmma_f32_16x16x32_bf16(false, a1, false, c3, (short)0, A13, false, false);
  }

  // scatter: C layout -> lane holds col n0; VGPR r holds row (r + half*8)
  const int n0 = lane & 15;
#pragma unroll
  for (int r = 0; r < 8; ++r) {
    int e0 = r + half * 8;
    int e1 = 16 + e0;
    if (ebase + e0 < E) {
      float* arow = agg + (size_t)dstL[wave][e0] * H_DIM;
      atomicAdd(arow + n0,      A00[r]);
      atomicAdd(arow + 16 + n0, A01[r]);
      atomicAdd(arow + 32 + n0, A02[r]);
      atomicAdd(arow + 48 + n0, A03[r]);
    }
    if (ebase + e1 < E) {
      float* arow = agg + (size_t)dstL[wave][e1] * H_DIM;
      atomicAdd(arow + n0,      A10[r]);
      atomicAdd(arow + 16 + n0, A11[r]);
      atomicAdd(arow + 32 + n0, A12[r]);
      atomicAdd(arow + 48 + n0, A13[r]);
    }
  }
}

// ----------------- self transform GEMM + combine ------------------
__global__ void __launch_bounds__(128, 1)
self_gemm_kernel(const unsigned short* __restrict__ hbf,
                 const unsigned short* __restrict__ Wsw, // layer slice
                 const float* __restrict__ self_b,
                 const float* __restrict__ deg,
                 float* __restrict__ agg, int N) {
  const int wave  = threadIdx.x >> 5;
  const int lane  = threadIdx.x & 31;
  const int tile  = blockIdx.x * (blockDim.x >> 5) + wave;
  const int mbase = tile * 16;
  if (mbase >= N) return;
  const int row  = lane & 15;
  const int half = lane >> 4;
  const int am   = (mbase + row < N) ? (mbase + row) : (N - 1);

  v8f acc0 = {}, acc1 = {}, acc2 = {}, acc3 = {};
#pragma unroll
  for (int kc = 0; kc < 2; ++kc) {
    const unsigned short* rp = hbf + (size_t)am * H_DIM + kc * 32 + half * 8;
    v16bf a;
#pragma unroll
    for (int j = 0; j < 8; ++j)  a[j]     = bits2bf(rp[j]);
#pragma unroll
    for (int j = 0; j < 8; ++j)  a[8 + j] = bits2bf(rp[16 + j]);
    const unsigned short* wp = Wsw + ((size_t)(kc * 4) * 32 + lane) * 16;
    v16bf b0 = load_frag(wp);
    v16bf b1 = load_frag(wp + 512);
    v16bf b2 = load_frag(wp + 1024);
    v16bf b3 = load_frag(wp + 1536);
    acc0 = __builtin_amdgcn_wmma_f32_16x16x32_bf16(false, a, false, b0, (short)0, acc0, false, false);
    acc1 = __builtin_amdgcn_wmma_f32_16x16x32_bf16(false, a, false, b1, (short)0, acc1, false, false);
    acc2 = __builtin_amdgcn_wmma_f32_16x16x32_bf16(false, a, false, b2, (short)0, acc2, false, false);
    acc3 = __builtin_amdgcn_wmma_f32_16x16x32_bf16(false, a, false, b3, (short)0, acc3, false, false);
  }
  const int n0 = lane & 15;
#pragma unroll
  for (int r = 0; r < 8; ++r) {
    int m = mbase + r + half * 8;
    if (m >= N) continue;
    float dg = fmaxf(deg[m], 1.0f);
    float* arow = agg + (size_t)m * H_DIM;
    arow[n0]      = arow[n0]      / dg + acc0[r] + self_b[n0];
    arow[16 + n0] = arow[16 + n0] / dg + acc1[r] + self_b[16 + n0];
    arow[32 + n0] = arow[32 + n0] / dg + acc2[r] + self_b[32 + n0];
    arow[48 + n0] = arow[48 + n0] / dg + acc3[r] + self_b[48 + n0];
  }
}

// --------------------------- BatchNorm ----------------------------
__global__ void bn_stats_kernel(const float* __restrict__ h2,
                                float* __restrict__ mu, float* __restrict__ var, int N) {
  int c = blockIdx.x;
  float s = 0.0f, s2 = 0.0f;
  for (int n = threadIdx.x; n < N; n += blockDim.x) {
    float v = h2[(size_t)n * H_DIM + c];
    s += v; s2 += v * v;
  }
  __shared__ float ls[256], lq[256];
  ls[threadIdx.x] = s; lq[threadIdx.x] = s2;
  __syncthreads();
  for (int off = 128; off > 0; off >>= 1) {
    if ((int)threadIdx.x < off) {
      ls[threadIdx.x] += ls[threadIdx.x + off];
      lq[threadIdx.x] += lq[threadIdx.x + off];
    }
    __syncthreads();
  }
  if (threadIdx.x == 0) {
    float m = ls[0] / (float)N;
    mu[c]  = m;
    var[c] = lq[0] / (float)N - m * m;   // biased variance
  }
}

__global__ void bn_apply_kernel(const float* __restrict__ h2,
                                const float* __restrict__ mu, const float* __restrict__ var,
                                const float* __restrict__ g, const float* __restrict__ b,
                                float* __restrict__ h, unsigned short* __restrict__ hbf, int N) {
  int idx = blockIdx.x * blockDim.x + threadIdx.x;
  if (idx >= N * H_DIM) return;
  int c = idx & 63;
  float v = g[c] * (h2[idx] - mu[c]) * rsqrtf(var[c] + BN_EPS) + b[c];
  v = fmaxf(v, 0.0f);
  h[idx]   = v;
  hbf[idx] = bf16bits(v);
}

// ---------------------------- pooling -----------------------------
__global__ void pool_kernel(const float* __restrict__ h, const int* __restrict__ batch,
                            float* __restrict__ psum, int* __restrict__ pmax,
                            float* __restrict__ pcnt, int N) {
  int idx = blockIdx.x * blockDim.x + threadIdx.x;
  if (idx >= N * H_DIM) return;
  int n = idx >> 6, c = idx & 63;
  int b = batch[n];
  float v = h[idx];
  atomicAdd(&psum[b * H_DIM + c], v);
  atomicMax(&pmax[b * H_DIM + c], __float_as_int(v));   // valid: v >= 0 post-ReLU
  if (c == 0) atomicAdd(&pcnt[b], 1.0f);
}

__global__ void classifier_kernel(const float* __restrict__ psum,
                                  const int* __restrict__ pmax,
                                  const float* __restrict__ pcnt,
                                  const float* __restrict__ cls_w,
                                  const float* __restrict__ cls_b,
                                  float* __restrict__ out) {
  int g = threadIdx.x;
  if (g >= NGRAPH) return;
  float cnt = fmaxf(pcnt[g], 1.0f);
  float acc = cls_b[0];
  for (int c = 0; c < H_DIM; ++c)
    acc += (psum[g * H_DIM + c] / cnt) * cls_w[c];
  for (int c = 0; c < H_DIM; ++c)
    acc += __int_as_float(pmax[g * H_DIM + c]) * cls_w[H_DIM + c];
  out[g] = 1.0f / (1.0f + expf(-acc));
}

// --------------------------- launcher -----------------------------
extern "C" void kernel_launch(void* const* d_in, const int* in_sizes, int n_in,
                              void* d_out, int out_size, void* d_ws, size_t ws_size,
                              hipStream_t stream) {
  (void)n_in; (void)out_size; (void)ws_size;
  const float* x      = (const float*)d_in[0];
  const int*   eidx   = (const int*)d_in[1];
  const int*   batch  = (const int*)d_in[2];
  const float* emb_w  = (const float*)d_in[3];
  const float* emb_b  = (const float*)d_in[4];
  const float* e1_w   = (const float*)d_in[5];
  const float* e1_b   = (const float*)d_in[6];
  const float* e2_w   = (const float*)d_in[7];
  const float* e2_b   = (const float*)d_in[8];
  const float* self_w = (const float*)d_in[9];
  const float* self_b = (const float*)d_in[10];
  const float* bn_g   = (const float*)d_in[11];
  const float* bn_b   = (const float*)d_in[12];
  const float* cls_w  = (const float*)d_in[13];
  const float* cls_b  = (const float*)d_in[14];
  float* out = (float*)d_out;

  const int N = in_sizes[2];
  const int E = in_sizes[1] / 2;
  const int* src = eidx;
  const int* dst = eidx + E;

  char* ws = (char*)d_ws;
  auto carve = [&](size_t bytes) {
    char* p = ws;
    ws += (bytes + 255) & ~(size_t)255;
    return p;
  };
  float*          deg  = (float*)carve((size_t)N * 4);
  float*          h    = (float*)carve((size_t)N * H_DIM * 4);
  unsigned short* hbf  = (unsigned short*)carve((size_t)N * H_DIM * 2);
  float*          agg  = (float*)carve((size_t)N * H_DIM * 4);
  float*          hid  = (float*)carve((size_t)E * EH_DIM * 4);
  unsigned short* Bsw  = (unsigned short*)carve((size_t)NLAYERS * NKCH * 4 * 32 * 16 * 2);
  unsigned short* Wsw  = (unsigned short*)carve((size_t)NLAYERS * 2 * 4 * 32 * 16 * 2);
  float*          mu   = (float*)carve(H_DIM * 4);
  float*          var  = (float*)carve(H_DIM * 4);
  float*          psum = (float*)carve(NGRAPH * H_DIM * 4);
  int*            pmax = (int*)carve(NGRAPH * H_DIM * 4);
  float*          pcnt = (float*)carve(NGRAPH * 4);

  const int TB = 256;
  // degrees (dst in-degree, clipped at use)
  zero_kernel<<<(N + TB - 1) / TB, TB, 0, stream>>>(deg, N);
  deg_kernel<<<(E + TB - 1) / TB, TB, 0, stream>>>(dst, deg, E);
  // pre-swizzled bf16 weights (all layers, once per launch)
  const int bswTot = NLAYERS * NKCH * 4 * 32 * 16;
  prep_bsw_kernel<<<(bswTot + TB - 1) / TB, TB, 0, stream>>>(e2_w, e2_b, Bsw, bswTot);
  const int wswTot = NLAYERS * 2 * 4 * 32 * 16;
  prep_wsw_kernel<<<(wswTot + TB - 1) / TB, TB, 0, stream>>>(self_w, Wsw, wswTot);
  // input embedding
  emb_kernel<<<(N * H_DIM + TB - 1) / TB, TB, 0, stream>>>(x, emb_w, emb_b, h, hbf, N);

  const int etiles = (E + EG_ROWS - 1) / EG_ROWS;
  const int mtiles = (N + 15) / 16;
  for (int l = 0; l < NLAYERS; ++l) {
    zero_kernel<<<(N * H_DIM + TB - 1) / TB, TB, 0, stream>>>(agg, N * H_DIM);
    hid_kernel<<<(E + TB - 1) / TB, TB, 0, stream>>>(
        h, src, dst, e1_w + l * EH_DIM, e1_b + l * EH_DIM, hid, E);
    edge_gemm_kernel<<<(etiles + EG_WAVES - 1) / EG_WAVES, EG_WAVES * 32, 0, stream>>>(
        h, hid, Bsw + (size_t)l * NKCH * 4 * 32 * 16, src, dst, agg, E);
    self_gemm_kernel<<<(mtiles + 3) / 4, 128, 0, stream>>>(
        hbf, Wsw + (size_t)l * 2 * 4 * 32 * 16, self_b + l * H_DIM, deg, agg, N);
    bn_stats_kernel<<<H_DIM, 256, 0, stream>>>(agg, mu, var, N);
    bn_apply_kernel<<<(N * H_DIM + TB - 1) / TB, TB, 0, stream>>>(
        agg, mu, var, bn_g + l * H_DIM, bn_b + l * H_DIM, h, hbf, N);
  }

  zero_kernel<<<(NGRAPH * H_DIM + TB - 1) / TB, TB, 0, stream>>>(psum, NGRAPH * H_DIM);
  zero_kernel<<<(NGRAPH * H_DIM + TB - 1) / TB, TB, 0, stream>>>((float*)pmax, NGRAPH * H_DIM);
  zero_kernel<<<1, 32, 0, stream>>>(pcnt, NGRAPH);
  pool_kernel<<<(N * H_DIM + TB - 1) / TB, TB, 0, stream>>>(h, batch, psum, pmax, pcnt, N);
  classifier_kernel<<<1, 32, 0, stream>>>(psum, pmax, pcnt, cls_w, cls_b, out);
}
